// WindowAttention_27470610826081
// MI455X (gfx1250) — compile-verified, matrix operand order checked by reading, not run on hardware
//
#include <hip/hip_runtime.h>

// ---------------------------------------------------------------------------
// Fused Swin window attention for MI455X (gfx1250, wave32, WMMA 16x16x32 f16)
// One workgroup (8 wave32) per window; all intermediates live in LDS.
// ---------------------------------------------------------------------------

typedef _Float16 v16h __attribute__((ext_vector_type(16)));
typedef float    v8f  __attribute__((ext_vector_type(8)));

#define NTOK     64
#define NHEADS   6
#define CDIM     192
#define HDIM     32
#define QK_SCALE 0.07216878364870323f   // 192^-0.5 (full-dim scale, as in source)

// LDS strides (in halfs / floats). All row strides keep 32-byte alignment so
// v16h (32 B) fragment loads are legal ds_load_b128 pairs.
#define XS  208   // Xh / Qh / Kh / Oh stride (192 + 16 pad) halfs
#define VTS 80    // Vt stride (64 tokens + 16 pad) halfs
#define PS  80    // probs stride halfs
// LDS byte offsets
#define QH_OFF   0
#define KH_OFF   26624            // 64*208*2
#define VT_OFF   53248            // + 64*208*2
#define RX_OFF   83968            // + 192*80*2   (Xh during phase 1; S+P later)
#define S_OFF    RX_OFF           // 64*64*4 = 16384 B
#define P_OFF    (RX_OFF + 16384) // 64*80*2 = 10240 B
#define OH_OFF   110592           // RX_OFF + 26624
#define LDS_BYTES 137216          // OH_OFF + 26624

// Fragment loader shared by A and (transposed-stored) B operands:
// lane l<16  -> row (row0+l),    halfs [k0, k0+16)
// lane l>=16 -> row (row0+l-16), halfs [k0+16, k0+32)
template <typename P>
__device__ __forceinline__ v16h load_frag(const P* p, int stride, int row0,
                                          int k0, int lane) {
  int r = row0 + (lane & 15);
  int k = k0 + ((lane >> 4) << 4);
  return *(const v16h*)(p + r * stride + k);
}

__device__ __forceinline__ v8f wmma_f16(v16h a, v16h b, v8f c) {
  return __builtin_amdgcn_wmma_f32_16x16x32_f16(false, a, false, b,
                                                (short)0, c, false, false);
}

// ---------------------------------------------------------------------------
// Prep: transpose weights to f16 (B-fragment friendly) and gather the
// relative-position bias table into a dense f32 [H][N][N] array.
// ---------------------------------------------------------------------------
__global__ __launch_bounds__(256) void win_attn_prep(
    const float* __restrict__ w_qkv,      // [192][576]
    const float* __restrict__ w_proj,     // [192][192]
    const float* __restrict__ bias_table, // [225][6]
    _Float16* __restrict__ wqkv_t,        // [576][192]
    _Float16* __restrict__ wproj_t,       // [192][192] (transposed)
    float* __restrict__ relbias)          // [6][64][64]
{
  int idx = blockIdx.x * 256 + threadIdx.x;
  if (idx < 576 * 192) {                       // Wqkv^T
    int n = idx / 192, k = idx % 192;
    wqkv_t[idx] = (_Float16)w_qkv[k * 576 + n];
  }
  int i2 = idx - 576 * 192;
  if (i2 >= 0 && i2 < 192 * 192) {             // Wproj^T
    int n = i2 / 192, k = i2 % 192;
    wproj_t[i2] = (_Float16)w_proj[k * 192 + n];
  }
  int i3 = idx - 576 * 192 - 192 * 192;
  if (i3 >= 0 && i3 < NHEADS * NTOK * NTOK) {  // gathered rel-pos bias
    int h = i3 >> 12, n = (i3 >> 6) & 63, m = i3 & 63;
    int dy = (n >> 3) - (m >> 3);
    int dx = (n & 7) - (m & 7);
    int ri = (dy + 7) * 15 + (dx + 7);
    relbias[i3] = bias_table[ri * NHEADS + h];
  }
}

// ---------------------------------------------------------------------------
// Main fused kernel: one window per block.
// ---------------------------------------------------------------------------
__global__ __launch_bounds__(256) void win_attn_main(
    const float*    __restrict__ x,        // [4096][64][192]
    const float*    __restrict__ mask,     // [64][64][64]
    const float*    __restrict__ b_qkv,    // [576]
    const float*    __restrict__ b_proj,   // [192]
    const _Float16* __restrict__ wqkv_t,   // [576][192]
    const _Float16* __restrict__ wproj_t,  // [192][192]
    const float*    __restrict__ relbias,  // [6][64][64]
    float*          __restrict__ out)      // [4096][64][192]
{
  extern __shared__ char smem[];
  _Float16* Qh = (_Float16*)(smem + QH_OFF);   // [64][XS]  head-concat cols
  _Float16* Kh = (_Float16*)(smem + KH_OFF);   // [64][XS]
  _Float16* Vt = (_Float16*)(smem + VT_OFF);   // [192][VTS] transposed V
  _Float16* Xh = (_Float16*)(smem + RX_OFF);   // [64][XS]  (phase 1 only)
  float*    S  = (float*)   (smem + S_OFF);    // [64][64]  per-head scores
  _Float16* Ph = (_Float16*)(smem + P_OFF);    // [64][PS]  per-head probs
  _Float16* Oh = (_Float16*)(smem + OH_OFF);   // [64][XS]  attn output

  const int t    = threadIdx.x;
  const int lane = t & 31;
  // Wave id as an SGPR: tile indices derived from it become scalar, so the
  // Q/K/V scatter branch is a scalar branch instead of an exec-mask dance.
  const int wvu  = __builtin_amdgcn_readfirstlane(t >> 5);   // 0..7
  const int blk  = blockIdx.x;
  const int win  = blk & 63;        // window index within image (for mask)
  const int rhi  = (lane >> 4) << 3;

  // Pull L2-resident weights toward this WGP early.
  if (t == 0) {
    __builtin_prefetch(wqkv_t, 0, 1);
    __builtin_prefetch(wproj_t, 0, 1);
  }

  // ---- stage X (f32 -> f16) into LDS -------------------------------------
  for (int e = t; e < NTOK * CDIM; e += 256) {
    int m = e / CDIM, c = e % CDIM;
    Xh[m * XS + c] = (_Float16)x[((size_t)blk * NTOK + m) * CDIM + c];
  }
  __syncthreads();

  // ---- phase 1: QKV = X * Wqkv + b  (64x576, K=192) ----------------------
  // 4x36 = 144 tiles of 16x16; 18 per wave.
  for (int i = 0; i < 18; ++i) {
    int tile = wvu + 8 * i;
    int tm = tile / 36, tn = tile % 36;      // scalar
    int nglob = tn * 16 + (lane & 15);
    float bq = b_qkv[nglob];
    v8f acc = {bq, bq, bq, bq, bq, bq, bq, bq};
#pragma unroll
    for (int kc = 0; kc < 6; ++kc) {
      v16h a = load_frag(Xh, XS, tm * 16, kc * 32, lane);
      v16h b = load_frag(wqkv_t, CDIM, tn * 16, kc * 32, lane);
      acc = wmma_f16(a, b, acc);
    }
    int part = tn / 12;              // 0=Q, 1=K, 2=V (scalar; tiles never straddle)
    int cc = nglob - part * CDIM;    // channel within part (head-concat)
#pragma unroll
    for (int r = 0; r < 8; ++r) {
      int m = tm * 16 + r + rhi;
      _Float16 hv = (_Float16)acc[r];
      if (part == 0)      Qh[m * XS + cc] = hv;
      else if (part == 1) Kh[m * XS + cc] = hv;
      else                Vt[cc * VTS + m] = hv;   // store V transposed
    }
  }
  __syncthreads();                   // Xh dead; S/Ph region live from here

  // ---- per-head attention ------------------------------------------------
  for (int h = 0; h < NHEADS; ++h) {
    // S = Q_h * K_h^T : 16 tiles, single K=32 WMMA each (2 per wave)
#pragma unroll
    for (int i = 0; i < 2; ++i) {
      int tile = wvu + 8 * i;
      int tm = tile >> 2, tn = tile & 3;
      v16h a = load_frag(Qh, XS, tm * 16, h * HDIM, lane);
      v16h b = load_frag(Kh, XS, tn * 16, h * HDIM, lane);
      v8f acc = {};
      acc = wmma_f16(a, b, acc);
#pragma unroll
      for (int r = 0; r < 8; ++r)
        S[(tm * 16 + r + rhi) * NTOK + tn * 16 + (lane & 15)] = acc[r];
    }
    __syncthreads();

    // softmax: 4 threads per row, 16 columns in registers per thread;
    // max/sum reduced across the 4-lane group with wave32 shuffles.
    {
      int row = t >> 2;
      int sub = t & 3;
      int base = sub * 16;
      const float* srow = S + row * NTOK + base;
      const float* bs = relbias + ((size_t)h * NTOK + row) * NTOK + base;
      const float* ms = mask + ((size_t)win * NTOK + row) * NTOK + base;
      float vbuf[16];
      float mx = -3.0e38f;
#pragma unroll
      for (int m = 0; m < 16; ++m) {
        float v = srow[m] * QK_SCALE + bs[m] + ms[m];
        vbuf[m] = v;
        mx = fmaxf(mx, v);
      }
      mx = fmaxf(mx, __shfl_xor(mx, 1, 32));
      mx = fmaxf(mx, __shfl_xor(mx, 2, 32));
      float sum = 0.f;
#pragma unroll
      for (int m = 0; m < 16; ++m) {
        float e = __expf(vbuf[m] - mx);
        vbuf[m] = e;
        sum += e;
      }
      sum += __shfl_xor(sum, 1, 32);
      sum += __shfl_xor(sum, 2, 32);
      float inv = 1.0f / sum;
#pragma unroll
      for (int m = 0; m < 16; ++m)
        Ph[row * PS + base + m] = (_Float16)(vbuf[m] * inv);
    }
    __syncthreads();

    // O_h = P * V_h : 8 tiles (one per wave), K=64 in 2 chunks
    {
      int tm = wvu >> 1, tn = wvu & 1;
      v8f acc = {};
#pragma unroll
      for (int kc = 0; kc < 2; ++kc) {
        v16h a = load_frag(Ph, PS, tm * 16, kc * 32, lane);
        v16h b = load_frag(Vt, VTS, h * HDIM + tn * 16, kc * 32, lane);
        acc = wmma_f16(a, b, acc);
      }
#pragma unroll
      for (int r = 0; r < 8; ++r) {
        int m = tm * 16 + r + rhi;
        Oh[m * XS + h * HDIM + tn * 16 + (lane & 15)] = (_Float16)acc[r];
      }
    }
    __syncthreads();   // S reused by next head; Ph reused after AV read
  }

  // ---- phase 5: out = O * Wproj + b_proj  (64x192, K=192) ----------------
  for (int i = 0; i < 6; ++i) {
    int tile = wvu + 8 * i;          // 0..47 (4 x 12 tiles)
    int tm = tile / 12, tn = tile % 12;
    int n = tn * 16 + (lane & 15);
    float bp = b_proj[n];
    v8f acc = {bp, bp, bp, bp, bp, bp, bp, bp};
#pragma unroll
    for (int kc = 0; kc < 6; ++kc) {
      v16h a = load_frag(Oh, XS, tm * 16, kc * 32, lane);
      v16h b = load_frag(wproj_t, CDIM, tn * 16, kc * 32, lane);
      acc = wmma_f16(a, b, acc);
    }
#pragma unroll
    for (int r = 0; r < 8; ++r) {
      int m = tm * 16 + r + rhi;
      out[((size_t)blk * NTOK + m) * CDIM + n] = acc[r];
    }
  }
}

// ---------------------------------------------------------------------------
extern "C" void kernel_launch(void* const* d_in, const int* in_sizes, int n_in,
                              void* d_out, int out_size, void* d_ws,
                              size_t ws_size, hipStream_t stream) {
  const float* x          = (const float*)d_in[0];
  const float* mask       = (const float*)d_in[1];
  const float* w_qkv      = (const float*)d_in[2];
  const float* b_qkv      = (const float*)d_in[3];
  const float* bias_table = (const float*)d_in[4];
  const float* w_proj     = (const float*)d_in[5];
  const float* b_proj     = (const float*)d_in[6];

  char* ws = (char*)d_ws;
  _Float16* wqkv_t  = (_Float16*)ws;                         // 221184 B
  _Float16* wproj_t = (_Float16*)(ws + 576 * 192 * 2);       //  73728 B
  float*    relbias = (float*)(ws + 576 * 192 * 2 + 192 * 192 * 2); // 98304 B

  // One-time weight transpose/convert + bias gather (re-run every call for
  // determinism; cheap: 172k elements).
  win_attn_prep<<<672, 256, 0, stream>>>(w_qkv, w_proj, bias_table,
                                         wqkv_t, wproj_t, relbias);

  win_attn_main<<<4096, 256, LDS_BYTES, stream>>>(
      x, mask, b_qkv, b_proj, wqkv_t, wproj_t, relbias, (float*)d_out);
}
